// EFLayoutGraphBlock_36661840838673
// MI455X (gfx1250) — matrix operand order repaired and checked by reading, not compiled, not run on hardware
//
#include <hip/hip_runtime.h>
#include <hip/hip_bf16.h>

typedef unsigned short u16;
typedef unsigned int u32;

typedef __attribute__((ext_vector_type(16))) __bf16 v16bf;
typedef __attribute__((ext_vector_type(8)))  float  v8f;

union FragBF { v16bf v; u32 u[8]; };

#define N_NODES 20000
#define N_EDGES 320000
#define ET      (N_EDGES + N_NODES)
#define IN_DIM  256
#define HC      256
#define HEADS   4
#define EDGE_DIM 16
#define NCAT    768      // W_l | W_r | W_res columns

#define A_STRIDE 264     // padded bf16 row stride for A tile (132 dwords -> banks m*4)
#define B_STRIDE 40      // padded bf16 row stride for B tile (20 dwords -> banks n*20)

// ---------- helpers ----------
__device__ __forceinline__ u16 f2bf(float f) {
    u32 u = __float_as_uint(f);
    u += 0x7FFFu + ((u >> 16) & 1u);   // round-to-nearest-even
    return (u16)(u >> 16);
}
__device__ __forceinline__ u32 encf(float f) {   // order-preserving f32 -> u32
    u32 u = __float_as_uint(f);
    return (u & 0x80000000u) ? ~u : (u | 0x80000000u);
}
__device__ __forceinline__ float decf(u32 e) {
    return (e & 0x80000000u) ? __uint_as_float(e & 0x7FFFFFFFu) : __uint_as_float(~e);
}

// ---------- fills ----------
__global__ void k_fill_f32(float* p, float v, int n) {
    int i = blockIdx.x * 256 + threadIdx.x;
    if (i < n) p[i] = v;
}
__global__ void k_fill_u32(u32* p, u32 v, int n) {
    int i = blockIdx.x * 256 + threadIdx.x;
    if (i < n) p[i] = v;
}

// ---------- self-loop attr: degree + segment-sum of edge_attr over dst ----------
__global__ void k_loop_stats(const int* dst, const float* edge_attr,
                             float* cnt, float* lsum) {
    int i = blockIdx.x * 256 + threadIdx.x;          // E*16 threads
    if (i >= N_EDGES * EDGE_DIM) return;
    int e = i >> 4, j = i & 15;
    int d = dst[e];
    if (j == 0) atomicAdd(cnt + d, 1.0f);
    atomicAdd(lsum + d * EDGE_DIM + j, edge_attr[i]);
}
__global__ void k_loop_fin(const float* cnt, const float* lsum, float* lattr) {
    int i = blockIdx.x * 256 + threadIdx.x;          // N*16 threads
    if (i >= N_NODES * EDGE_DIM) return;
    lattr[i] = lsum[i] / fmaxf(cnt[i >> 4], 1.0f);
}

// ---------- bf16 conversions ----------
__global__ void k_cvt_x(const float* x, u16* xbf, int n) {
    int i = blockIdx.x * 256 + threadIdx.x;
    if (i < n) xbf[i] = f2bf(x[i]);
}
// transposed weight: wbfT[c][k], c in [0,768), k in [0,256) -- contiguous K per column
__global__ void k_cvt_wT(const float* Wl, const float* Wr, const float* Wres, u16* wbfT) {
    int i = blockIdx.x * 256 + threadIdx.x;          // 768*256 threads
    if (i >= NCAT * IN_DIM) return;
    int c = i >> 8;          // output column 0..767
    int k = i & 255;         // K index
    const float* Wsrc; int cc;
    if (c < 256)      { Wsrc = Wl;   cc = c; }
    else if (c < 512) { Wsrc = Wr;   cc = c - 256; }
    else              { Wsrc = Wres; cc = c - 512; }
    wbfT[i] = f2bf(Wsrc[k * 256 + cc]);
}

// ---------- fused WMMA GEMM: [x_l | x_r | res] = x(bf16) @ Wcat(bf16) + bias ----------
// block: 256 thr = 8 waves; tile 16(M) x 128(N); grid (1250, 6)
__global__ __launch_bounds__(256) void k_gemm_wmma(
    const u16* __restrict__ xbf, const u16* __restrict__ wbfT,
    const float* __restrict__ b_l, const float* __restrict__ b_r,
    float* __restrict__ xl, float* __restrict__ xr, float* __restrict__ res) {
    __shared__ u16 Asm[16 * A_STRIDE];    // [m][k], full K panel, padded rows
    __shared__ u16 Bsm[128 * B_STRIDE];   // [n][k] K-slice (k-contiguous), padded rows

    const int tid  = threadIdx.x;
    const int wave = tid >> 5, lane = tid & 31;
    const int row0 = blockIdx.x * 16;
    const int col0 = blockIdx.y * 128;

    // cooperative A tile load: 16x256 bf16; thread t copies 32B of row (t>>4)
    {
        const uint4* xg = (const uint4*)(xbf + (size_t)row0 * 256);
        uint4* Au4 = (uint4*)Asm;
        // global: row (t>>4), dword offset (t&15)*8 ; LDS row stride 132 dw = 33 uint4
        int r = tid >> 4, cdw4 = (tid & 15) * 2;     // uint4 units: 2 per thread
        Au4[r * 33 + cdw4 + 0] = xg[r * 32 + cdw4 + 0];
        Au4[r * 33 + cdw4 + 1] = xg[r * 32 + cdw4 + 1];
    }

    const int m  = lane & 15;
    const int kh = lane >> 4;
    const int nl = wave * 16 + (lane & 15);   // local output column 0..127

    v8f c = {};
#pragma unroll
    for (int k0 = 0; k0 < 256; k0 += 32) {
        // stage B slice: Bsm[n][kk], all b128 copies (wbfT rows are k-contiguous)
        {
            int n = tid >> 1, half = tid & 1;        // 128 n x 2 halves of 16 bf16
            const uint4* wg = (const uint4*)(wbfT + (size_t)(col0 + n) * 256 + k0 + half * 16);
            uint4* B4 = (uint4*)(Bsm + n * B_STRIDE + half * 16);
            B4[0] = wg[0];
            B4[1] = wg[1];
        }
        __syncthreads();

        FragBF af, bfr;
        const u32* Au = (const u32*)Asm;
        const u32* Bu = (const u32*)Bsm;
#pragma unroll
        for (int v = 0; v < 8; ++v) {
            // A operand, contiguous k order used here (matches B): k = kh*16 + 2v (+pair)
            int ka = (kh << 4) + (v << 1);
            af.u[v] = Au[(m * A_STRIDE + k0 + ka) >> 1];
            // B operand, same contiguous k order as A: k = kh*16 + 2v (+pair), column = lane&15
            bfr.u[v] = Bu[nl * (B_STRIDE / 2) + kh * 8 + v];
        }
        c = __builtin_amdgcn_wmma_f32_16x16x32_bf16(
                false, af.v, false, bfr.v, (short)0, c, false, false);
        __syncthreads();
    }

    // writeback with biases; route column to x_l / x_r / res
    const int nc = col0 + nl;
    float bias; float* dstp; int cc;
    if (nc < 256)      { bias = b_l[nc];        dstp = xl;  cc = nc; }
    else if (nc < 512) { bias = b_r[nc - 256];  dstp = xr;  cc = nc - 256; }
    else               { bias = 0.0f;           dstp = res; cc = nc - 512; }
#pragma unroll
    for (int r = 0; r < 8; ++r) {
        int row = row0 + kh * 8 + r;           // C layout: VGPR r -> M=r / r+8
        dstp[(size_t)row * 256 + cc] = c[r] + bias;
    }
}

// ---------- edge attention logits: 1 wave per edge, 8 channels per lane ----------
__global__ __launch_bounds__(256) void k_edge_logits(
    const int* __restrict__ src, const int* __restrict__ dst,
    const float* __restrict__ edge_attr, const float* __restrict__ lattr,
    const float* __restrict__ We, const float* __restrict__ att,
    const float* __restrict__ xl, const float* __restrict__ xr,
    float* __restrict__ logits, u32* __restrict__ menc) {
    __shared__ float We_s[EDGE_DIM * 256];   // 16 KB
    __shared__ float att_s[256];
    const int tid = threadIdx.x;
#pragma unroll
    for (int i = 0; i < EDGE_DIM; ++i) We_s[tid + i * 256] = We[tid + i * 256];
    att_s[tid] = att[tid];
    __syncthreads();

    const int e = blockIdx.x * 8 + (tid >> 5);   // Et = 340000 = 8*42500 exactly
    const int lane = tid & 31;
    int s, d; const float* ea;
    if (e < N_EDGES) { s = src[e]; d = dst[e]; ea = edge_attr + (size_t)e * EDGE_DIM; }
    else             { s = e - N_EDGES; d = s;  ea = lattr + (size_t)s * EDGE_DIM; }

    const int c0 = lane * 8;
    const float* xls = xl + (size_t)s * 256 + c0;
    const float* xrd = xr + (size_t)d * 256 + c0;

    float eav[EDGE_DIM];
#pragma unroll
    for (int k = 0; k < EDGE_DIM; ++k) eav[k] = ea[k];

    float p = 0.0f;
#pragma unroll
    for (int j = 0; j < 8; ++j) {
        float v = xls[j] + xrd[j];
#pragma unroll
        for (int k = 0; k < EDGE_DIM; ++k) v += eav[k] * We_s[k * 256 + c0 + j];
        v = fmaxf(v, 0.2f * v);               // LeakyReLU(0.2)
        p += v * att_s[c0 + j];
    }
    // reduce 8 lanes of the same head (lanes l..l+7, head = lane/8)
    p += __shfl_xor(p, 1, 32);
    p += __shfl_xor(p, 2, 32);
    p += __shfl_xor(p, 4, 32);
    if ((lane & 7) == 0) {
        int h = lane >> 3;
        logits[(size_t)e * HEADS + h] = p;
        atomicMax(menc + (size_t)d * HEADS + h, encf(p));
    }
}

// ---------- exp(logit - max) and segment denominator ----------
__global__ void k_edge_exp(const int* __restrict__ dst, const u32* __restrict__ menc,
                           float* __restrict__ logits, float* __restrict__ denom) {
    int i = blockIdx.x * 256 + threadIdx.x;     // Et*4 threads
    if (i >= ET * HEADS) return;
    int e = i >> 2, h = i & 3;
    int d = (e < N_EDGES) ? dst[e] : (e - N_EDGES);
    float a = __expf(logits[i] - decf(menc[d * HEADS + h]));
    logits[i] = a;                               // reuse buffer for exp values
    atomicAdd(denom + d * HEADS + h, a);
}

// ---------- weighted aggregation: out[dst] += alpha * x_l[src] ----------
__global__ __launch_bounds__(256) void k_aggregate(
    const int* __restrict__ src, const int* __restrict__ dst,
    const float* __restrict__ xl, const float* __restrict__ aexp,
    const float* __restrict__ denom, float* __restrict__ out) {
    const int tid = threadIdx.x;
    const int e = blockIdx.x * 8 + (tid >> 5);
    const int lane = tid & 31;
    int s, d;
    if (e < N_EDGES) { s = src[e]; d = dst[e]; }
    else             { s = e - N_EDGES; d = s; }
    const int h = lane >> 3;
    const float alpha = aexp[(size_t)e * HEADS + h] / denom[(size_t)d * HEADS + h];
    const int c0 = lane * 8;
    const float* xls = xl + (size_t)s * 256 + c0;
    float* op = out + (size_t)d * 256 + c0;
#pragma unroll
    for (int j = 0; j < 8; ++j) atomicAdd(op + j, alpha * xls[j]);
}

// ---------- residual + bias + ELU ----------
__global__ void k_finalize(float* __restrict__ out, const float* __restrict__ res,
                           const float* __restrict__ bias, int n) {
    int i = blockIdx.x * 256 + threadIdx.x;
    if (i >= n) return;
    float v = out[i] + res[i] + bias[i & 255];
    out[i] = (v > 0.0f) ? v : (__expf(v) - 1.0f);
}

// =====================================================================
extern "C" void kernel_launch(void* const* d_in, const int* in_sizes, int n_in,
                              void* d_out, int out_size, void* d_ws, size_t ws_size,
                              hipStream_t stream) {
    const float* x        = (const float*)d_in[0];
    const int*   eidx     = (const int*)d_in[1];          // [2, E] int32
    const int*   src      = eidx;
    const int*   dst      = eidx + N_EDGES;
    const float* edge_att = (const float*)d_in[2];
    const float* W_l      = (const float*)d_in[3];
    const float* b_l      = (const float*)d_in[4];
    const float* W_r      = (const float*)d_in[5];
    const float* b_r      = (const float*)d_in[6];
    const float* W_e      = (const float*)d_in[7];
    const float* att      = (const float*)d_in[8];
    const float* W_res    = (const float*)d_in[9];
    const float* bias     = (const float*)d_in[10];
    float* out = (float*)d_out;

    // workspace carve-up (256B aligned)
    char* ws = (char*)d_ws; size_t off = 0;
    auto carve = [&](size_t bytes) -> void* {
        void* p = ws + off; off = (off + bytes + 255) & ~(size_t)255; return p;
    };
    u16*   xbf   = (u16*)  carve((size_t)N_NODES * IN_DIM * 2);
    u16*   wbfT  = (u16*)  carve((size_t)NCAT * IN_DIM * 2);
    float* xl    = (float*)carve((size_t)N_NODES * HC * 4);
    float* xr    = (float*)carve((size_t)N_NODES * HC * 4);
    float* resb  = (float*)carve((size_t)N_NODES * HC * 4);
    float* cnt   = (float*)carve((size_t)N_NODES * 4);
    float* lsum  = (float*)carve((size_t)N_NODES * EDGE_DIM * 4);
    float* lattr = (float*)carve((size_t)N_NODES * EDGE_DIM * 4);
    float* logit = (float*)carve((size_t)ET * HEADS * 4);
    u32*   menc  = (u32*)  carve((size_t)N_NODES * HEADS * 4);
    float* denom = (float*)carve((size_t)N_NODES * HEADS * 4);
    (void)ws_size; (void)n_in; (void)in_sizes; (void)out_size;

    auto blk = [](int n) { return (n + 255) / 256; };

    // 1) init accumulators
    k_fill_f32<<<blk(N_NODES * HC), 256, 0, stream>>>(out, 0.0f, N_NODES * HC);
    k_fill_f32<<<blk(N_NODES), 256, 0, stream>>>(cnt, 0.0f, N_NODES);
    k_fill_f32<<<blk(N_NODES * EDGE_DIM), 256, 0, stream>>>(lsum, 0.0f, N_NODES * EDGE_DIM);
    k_fill_f32<<<blk(N_NODES * HEADS), 256, 0, stream>>>(denom, 0.0f, N_NODES * HEADS);
    k_fill_u32<<<blk(N_NODES * HEADS), 256, 0, stream>>>(menc, 0x007FFFFFu /* enc(-inf) */,
                                                         N_NODES * HEADS);
    // 2) self-loop mean edge attr
    k_loop_stats<<<blk(N_EDGES * EDGE_DIM), 256, 0, stream>>>(dst, edge_att, cnt, lsum);
    k_loop_fin<<<blk(N_NODES * EDGE_DIM), 256, 0, stream>>>(cnt, lsum, lattr);

    // 3) bf16 conversions for the WMMA GEMM (weights pre-transposed, k-contiguous)
    k_cvt_x<<<blk(N_NODES * IN_DIM), 256, 0, stream>>>(x, xbf, N_NODES * IN_DIM);
    k_cvt_wT<<<blk(NCAT * IN_DIM), 256, 0, stream>>>(W_l, W_r, W_res, wbfT);

    // 4) fused node GEMM: x_l / x_r / res
    dim3 ggrid(N_NODES / 16, NCAT / 128);
    k_gemm_wmma<<<ggrid, 256, 0, stream>>>(xbf, wbfT, b_l, b_r, xl, xr, resb);

    // 5) per-edge attention logits + segment max (Et = 8 * 42500)
    k_edge_logits<<<ET / 8, 256, 0, stream>>>(src, dst, edge_att, lattr, W_e, att,
                                              xl, xr, logit, menc);
    // 6) exp + denom
    k_edge_exp<<<blk(ET * HEADS), 256, 0, stream>>>(dst, menc, logit, denom);
    // 7) aggregate
    k_aggregate<<<ET / 8, 256, 0, stream>>>(src, dst, xl, logit, denom, out);
    // 8) residual + bias + ELU
    k_finalize<<<blk(N_NODES * HC), 256, 0, stream>>>(out, resb, bias, N_NODES * HC);
}
